// Attention_60576218743061
// MI455X (gfx1250) — matrix operand order repaired and checked by reading, not compiled, not run on hardware
//
#include <hip/hip_runtime.h>
#include <hip/hip_bf16.h>

// ---------------------------------------------------------------------------
// MI455X (gfx1250) multi-head attention forward.
//   B=2, N=2048, DIM=1024, HEADS=16, DIM_HEAD=64
// Pipeline:
//   1) qkv_proj_kernel : x @ {Wq,Wk,Wv} -> bf16 Q (pre-scaled), K [b,h,n,64],
//                        V transposed [b,h,64,n]   (WMMA bf16, LDS-tiled GEMM)
//   2) attn_kernel     : flash-attention; K/V tiles double-buffered in LDS via
//                        global_load_async_to_lds_b128 (ASYNCcnt credits:
//                        s_wait_asynccnt 4 while the next tile is in flight),
//                        online softmax, P->A transpose via per-wave LDS +
//                        s_wait_dscnt, WMMA bf16 for QK^T and PV
//   3) out_proj_kernel : O @ Wo + bo -> f32 output (async-staged A tile)
// Workspace: 4 segments x 4,194,304 bf16 = 32 MB.
// ---------------------------------------------------------------------------

typedef __attribute__((ext_vector_type(16))) __bf16 v16bf;
typedef __attribute__((ext_vector_type(8)))  float  v8f;

#define ATTN_B    2
#define ATTN_N    2048
#define ATTN_DIM  1024
#define ATTN_H    16
#define ATTN_D    64
#define QK_SCALE  0.125f        // 64^-0.5
#define NEG_BIG   (-3.0e38f)

__device__ __forceinline__ unsigned short f2bf(float f) {
  union { float f; unsigned int u; } c; c.f = f;
  unsigned int u = c.u;
  u += 0x7FFFu + ((u >> 16) & 1u);          // round-to-nearest-even
  return (unsigned short)(u >> 16);
}
__device__ __forceinline__ unsigned int pack2(float a, float b) {
  return (unsigned int)f2bf(a) | ((unsigned int)f2bf(b) << 16);
}

union Frag { v16bf v; unsigned int u[8]; };

__device__ __forceinline__ v8f wmma_bf16(v16bf a, v16bf b, v8f c) {
  return __builtin_amdgcn_wmma_f32_16x16x32_bf16(false, a, false, b,
                                                 (short)0, c, false, false);
}
// A-matrix (16x32, 16-bit) K index of packed-pair slot s (lane half adds +8):
//   slot 0..3 -> K = 0,2,4,6 ; slot 4..7 -> K = 16,18,20,22
__device__ __forceinline__ int a_ksel(int s) { return (s < 4) ? 2 * s : 8 + 2 * s; }

// Async copy of 16 bytes global -> LDS (GVS mode), tracked by ASYNCcnt.
__device__ __forceinline__ void async_copy_b128(unsigned int lds_byte_addr,
                                                unsigned int glb_byte_off,
                                                const void* glb_base) {
  asm volatile("global_load_async_to_lds_b128 %0, %1, %2"
               :: "v"(lds_byte_addr), "v"(glb_byte_off), "s"(glb_base)
               : "memory");
}
__device__ __forceinline__ void wait_async0() {
  asm volatile("s_wait_asynccnt 0x0" ::: "memory");
}
__device__ __forceinline__ void wait_async4() {  // current tile done; next in flight
  asm volatile("s_wait_asynccnt 0x4" ::: "memory");
}
// Low 32 bits of a generic pointer to LDS == group-segment byte offset.
__device__ __forceinline__ unsigned int lds_addr(const void* p) {
  return (unsigned int)(size_t)p;
}

// ---------------------------------------------------------------------------
// Kernel 1: fused QKV projection.  Grid: (3072/64, 4096/64), 128 threads.
// ---------------------------------------------------------------------------
__global__ __launch_bounds__(128) void qkv_proj_kernel(
    const float* __restrict__ x,  const float* __restrict__ Wq,
    const float* __restrict__ Wk, const float* __restrict__ Wv,
    unsigned short* __restrict__ qo, unsigned short* __restrict__ ko,
    unsigned short* __restrict__ vto)
{
  __shared__ unsigned int   As[64][17];   // [m][k/2] bf16 pairs, padded stride
  __shared__ unsigned short Bs[64][34];   // [n][k]   bf16, padded stride

  const int tid  = threadIdx.x;
  const int lane = tid & 31, wave = tid >> 5;
  const int hi = lane >> 4, ln = lane & 15;
  const int row0 = blockIdx.y * 64;
  const int cg   = blockIdx.x * 64;       // global output column (0..3071)
  const int mtx  = cg >> 10;              // 0=Q 1=K 2=V
  const int c0   = cg & 1023;
  const float* __restrict__ W = (mtx == 0) ? Wq : ((mtx == 1) ? Wk : Wv);

  v8f acc[4];
#pragma unroll
  for (int t = 0; t < 4; ++t)
#pragma unroll
    for (int i = 0; i < 8; ++i) acc[t][i] = 0.f;

  for (int k0 = 0; k0 < ATTN_DIM; k0 += 32) {
    { // stage A tile 64x32 (f32 -> packed bf16)
      int m = tid >> 1, kb = (tid & 1) * 16;
      const float4* src = (const float4*)(x + (size_t)(row0 + m) * ATTN_DIM + k0 + kb);
      if (k0 + 32 < ATTN_DIM)
        __builtin_prefetch(x + (size_t)(row0 + m) * ATTN_DIM + k0 + 32 + kb, 0, 1);
#pragma unroll
      for (int j = 0; j < 4; ++j) {
        float4 f = src[j];
        As[m][(kb >> 1) + 2 * j    ] = pack2(f.x, f.y);
        As[m][(kb >> 1) + 2 * j + 1] = pack2(f.z, f.w);
      }
    }
    { // stage B tile 32x64, transposed to [n][k] so K-pairs are contiguous
      int kr = tid >> 2, nb = (tid & 3) * 16;
      const float4* src = (const float4*)(W + (size_t)(k0 + kr) * ATTN_DIM + c0 + nb);
      if (k0 + 32 < ATTN_DIM)
        __builtin_prefetch(W + (size_t)(k0 + 32 + kr) * ATTN_DIM + c0 + nb, 0, 1);
#pragma unroll
      for (int j = 0; j < 4; ++j) {
        float4 f = src[j];
        Bs[nb + 4 * j    ][kr] = f2bf(f.x);
        Bs[nb + 4 * j + 1][kr] = f2bf(f.y);
        Bs[nb + 4 * j + 2][kr] = f2bf(f.z);
        Bs[nb + 4 * j + 3][kr] = f2bf(f.w);
      }
    }
    __syncthreads();

    Frag a;
#pragma unroll
    for (int s = 0; s < 8; ++s)
      a.u[s] = As[wave * 16 + ln][(a_ksel(s) + hi * 8) >> 1];
#pragma unroll
    for (int t = 0; t < 4; ++t) {
      Frag b;
#pragma unroll
      for (int s = 0; s < 8; ++s)
        b.u[s] = *(const unsigned int*)&Bs[t * 16 + ln][a_ksel(s) + hi * 8];
      acc[t] = wmma_bf16(a.v, b.v, acc[t]);
    }
    __syncthreads();
  }

  // epilogue: scatter into attention-friendly layouts
#pragma unroll
  for (int t = 0; t < 4; ++t)
#pragma unroll
    for (int r = 0; r < 8; ++r) {
      int grow = row0 + wave * 16 + r + hi * 8;   // 0..4095  (b*2048+n)
      int gcol = c0 + t * 16 + ln;                // 0..1023  (h*64+d)
      int b = grow >> 11, n = grow & 2047;
      int h = gcol >> 6,  d = gcol & 63;
      float val = acc[t][r];
      if (mtx == 0)
        qo[((size_t)(b * ATTN_H + h) * ATTN_N + n) * ATTN_D + d] = f2bf(val * QK_SCALE);
      else if (mtx == 1)
        ko[((size_t)(b * ATTN_H + h) * ATTN_N + n) * ATTN_D + d] = f2bf(val);
      else  // V stored transposed [b,h,d,n] so PV B-fragments are pair-contiguous
        vto[((size_t)(b * ATTN_H + h) * ATTN_D + d) * ATTN_N + n] = f2bf(val);
    }
}

// ---------------------------------------------------------------------------
// Kernel 2: flash attention.  Grid: (2048/64, 16, 2), 128 threads (4 waves),
// each wave owns 16 query rows; K/V tiles double-buffered in LDS via async
// copies shared by all 4 waves.  Each tile-stage issues exactly 4 async ops
// per wave, so s_wait_asynccnt 4 retires the current tile while the next
// stays in flight.  LDS row strides are 16B multiples (B128 alignment) with
// odd 16B-chunk counts (9 / 5) so 16-lane fragment reads are conflict-free.
// ---------------------------------------------------------------------------
__device__ __forceinline__ void stage_kv(int tid, int j0,
    const unsigned short* kh, const unsigned short* vh,
    unsigned short (*Ksb)[72], unsigned short (*Vsb)[40])
{
#pragma unroll
  for (int i = 0; i < 2; ++i) {        // K tile 32x64 bf16: 256 B128 copies
    int idx = i * 128 + tid;
    int row = idx >> 3, ch = idx & 7;
    async_copy_b128(lds_addr(&Ksb[row][ch * 8]),
                    (unsigned int)(((j0 + row) * ATTN_D + ch * 8) * 2), kh);
  }
#pragma unroll
  for (int i = 0; i < 2; ++i) {        // V tile 64x32 bf16: 256 B128 copies
    int idx = i * 128 + tid;
    int row = idx >> 2, ch = idx & 3;
    async_copy_b128(lds_addr(&Vsb[row][ch * 8]),
                    (unsigned int)((row * ATTN_N + j0 + ch * 8) * 2), vh);
  }
}

__global__ __launch_bounds__(128) void attn_kernel(
    const unsigned short* __restrict__ q, const unsigned short* __restrict__ k,
    const unsigned short* __restrict__ vt, const int* __restrict__ mask,
    unsigned short* __restrict__ o)
{
  __shared__ unsigned short Ks[2][32][72];  // K tiles  [key][d]  (144B rows)
  __shared__ unsigned short Vs[2][64][40];  // V^T tiles [d][key] (80B rows)
  __shared__ unsigned short Pl[4][16][40];  // per-wave P bounce buffer

  const int tid = threadIdx.x;
  const int lane = tid & 31, wave = tid >> 5;
  const int hi = lane >> 4, ln = lane & 15;
  const int b = blockIdx.z, h = blockIdx.y;
  const int i0 = blockIdx.x * 64 + wave * 16;

  const unsigned short* qh = q  + (size_t)(b * ATTN_H + h) * ATTN_N * ATTN_D;
  const unsigned short* kh = k  + (size_t)(b * ATTN_H + h) * ATTN_N * ATTN_D;
  const unsigned short* vh = vt + (size_t)(b * ATTN_H + h) * ATTN_D * ATTN_N;
  const int* mb = mask + (size_t)b * ATTN_N * ATTN_N;

  // prologue: stage first K/V tile while we load Q fragments
  stage_kv(tid, 0, kh, vh, Ks[0], Vs[0]);

  // resident Q fragments (16 rows x 64 d) = two 16x32 A-frags, pre-scaled
  Frag qa[2];
#pragma unroll
  for (int f = 0; f < 2; ++f)
#pragma unroll
    for (int s = 0; s < 8; ++s)
      qa[f].u[s] = *(const unsigned int*)(qh + (size_t)(i0 + ln) * ATTN_D
                                          + f * 32 + a_ksel(s) + hi * 8);

  v8f oacc[4];
#pragma unroll
  for (int t = 0; t < 4; ++t)
#pragma unroll
    for (int i = 0; i < 8; ++i) oacc[t][i] = 0.f;
  float mrow[8], lrow[8];
#pragma unroll
  for (int r = 0; r < 8; ++r) { mrow[r] = NEG_BIG; lrow[r] = 0.f; }

  int cur = 0;
  for (int j0 = 0; j0 < ATTN_N; j0 += 32) {
    const bool has_next = (j0 + 32 < ATTN_N);
    // issue next tile's copies into the back buffer, then retire current tile
    if (has_next) {
      stage_kv(tid, j0 + 32, kh, vh, Ks[cur ^ 1], Vs[cur ^ 1]);
      wait_async4();                   // 4 outstanding = next tile only
    } else {
      wait_async0();
    }
    if (has_next)                      // prefetch next mask row segment
      __builtin_prefetch(mb + (size_t)(i0 + hi * 8) * ATTN_N + j0 + 32 + ln, 0, 1);
    __syncthreads();                   // all waves' current-tile copies visible

    // ---- S = Q K^T  (two 16x16 C tiles covering 32 keys) ----
    v8f s0, s1;
#pragma unroll
    for (int i = 0; i < 8; ++i) { s0[i] = 0.f; s1[i] = 0.f; }
#pragma unroll
    for (int f = 0; f < 2; ++f) {
      Frag bk0, bk1;
#pragma unroll
      for (int sv = 0; sv < 8; ++sv) {
        int kd = f * 32 + a_ksel(sv) + hi * 8;   // d index (B-matrix row)
        bk0.u[sv] = *(const unsigned int*)&Ks[cur][ln][kd];
        bk1.u[sv] = *(const unsigned int*)&Ks[cur][16 + ln][kd];
      }
      s0 = wmma_bf16(qa[f].v, bk0.v, s0);
      s1 = wmma_bf16(qa[f].v, bk1.v, s1);
    }

    // ---- mask (keep where mask==0) + row max, in C layout ----
    float mnew[8];
#pragma unroll
    for (int r = 0; r < 8; ++r) {
      int gi = i0 + r + hi * 8;
      s0[r] = (mb[(size_t)gi * ATTN_N + j0 + ln]      != 0) ? NEG_BIG : s0[r];
      s1[r] = (mb[(size_t)gi * ATTN_N + j0 + 16 + ln] != 0) ? NEG_BIG : s1[r];
      mnew[r] = fmaxf(mrow[r], fmaxf(s0[r], s1[r]));
    }
#pragma unroll
    for (int off = 8; off >= 1; off >>= 1)
#pragma unroll
      for (int r = 0; r < 8; ++r)
        mnew[r] = fmaxf(mnew[r], __shfl_xor(mnew[r], off, 32));

    // ---- online softmax update ----
    float alpha[8], psum[8];
#pragma unroll
    for (int r = 0; r < 8; ++r) {
      alpha[r] = __expf(mrow[r] - mnew[r]);
      mrow[r]  = mnew[r];
      float p0 = __expf(s0[r] - mnew[r]);
      float p1 = __expf(s1[r] - mnew[r]);
      s0[r] = p0; s1[r] = p1;
      psum[r] = p0 + p1;
    }
#pragma unroll
    for (int off = 8; off >= 1; off >>= 1)
#pragma unroll
      for (int r = 0; r < 8; ++r)
        psum[r] += __shfl_xor(psum[r], off, 32);
#pragma unroll
    for (int r = 0; r < 8; ++r) lrow[r] = lrow[r] * alpha[r] + psum[r];
#pragma unroll
    for (int t = 0; t < 4; ++t)
#pragma unroll
      for (int r = 0; r < 8; ++r) oacc[t][r] *= alpha[r];

    // ---- P: C layout -> A layout via per-wave LDS bounce ----
#pragma unroll
    for (int r = 0; r < 8; ++r) {
      Pl[wave][r + hi * 8][ln]      = f2bf(s0[r]);
      Pl[wave][r + hi * 8][16 + ln] = f2bf(s1[r]);
    }
    asm volatile("s_wait_dscnt 0" ::: "memory");  // DS RAW, same wave (in-order)
    Frag pa;
#pragma unroll
    for (int sv = 0; sv < 8; ++sv)
      pa.u[sv] = *(const unsigned int*)&Pl[wave][ln][a_ksel(sv) + hi * 8];

    // ---- O += P V  (V tile from LDS, pair-contiguous along keys) ----
#pragma unroll
    for (int t = 0; t < 4; ++t) {
      Frag vb;
#pragma unroll
      for (int sv = 0; sv < 8; ++sv)
        vb.u[sv] = *(const unsigned int*)&Vs[cur][t * 16 + ln][a_ksel(sv) + hi * 8];
      oacc[t] = wmma_bf16(pa.v, vb.v, oacc[t]);
    }
    __syncthreads();   // all reads of [cur] done before it is restaged
    cur ^= 1;
  }

  // ---- normalize and emit O as bf16 [b*N][H*D] for the output projection ----
#pragma unroll
  for (int r = 0; r < 8; ++r) {
    float inv = (lrow[r] > 0.f) ? (1.f / lrow[r]) : 0.f;
#pragma unroll
    for (int t = 0; t < 4; ++t) {
      int n   = i0 + r + hi * 8;
      int col = h * ATTN_D + t * 16 + ln;
      o[(size_t)(b * ATTN_N + n) * (ATTN_H * ATTN_D) + col] = f2bf(oacc[t][r] * inv);
    }
  }
}

// ---------------------------------------------------------------------------
// Kernel 3: output projection O @ Wo + bo.  Grid: (1024/64, 4096/64), 128 thr.
// A tile (already bf16) staged via async LDS copies, overlapping the VALU
// f32->bf16 conversion of the B tile.
// ---------------------------------------------------------------------------
__global__ __launch_bounds__(128) void out_proj_kernel(
    const unsigned short* __restrict__ A, const float* __restrict__ Wo,
    const float* __restrict__ bo, float* __restrict__ out)
{
  __shared__ unsigned short As2[64][40];  // [m][k] bf16, 80B rows (16B aligned)
  __shared__ unsigned short Bs[64][34];   // [n][k] bf16, padded stride

  const int tid  = threadIdx.x;
  const int lane = tid & 31, wave = tid >> 5;
  const int hi = lane >> 4, ln = lane & 15;
  const int row0 = blockIdx.y * 64;
  const int c0   = blockIdx.x * 64;

  v8f acc[4];
#pragma unroll
  for (int t = 0; t < 4; ++t)
#pragma unroll
    for (int i = 0; i < 8; ++i) acc[t][i] = 0.f;

  for (int k0 = 0; k0 < ATTN_DIM; k0 += 32) {
    // async-stage A tile 64x32 bf16 (pure byte copy)
#pragma unroll
    for (int i = 0; i < 2; ++i) {
      int idx = i * 128 + tid;
      int row = idx >> 2, ch = idx & 3;
      async_copy_b128(lds_addr(&As2[row][ch * 8]),
                      (unsigned int)((((row0 + row) * ATTN_DIM) + k0 + ch * 8) * 2), A);
    }
    { // B tile f32 -> bf16, transposed to [n][k]
      int kr = tid >> 2, nb = (tid & 3) * 16;
      const float4* src = (const float4*)(Wo + (size_t)(k0 + kr) * ATTN_DIM + c0 + nb);
      if (k0 + 32 < ATTN_DIM)
        __builtin_prefetch(Wo + (size_t)(k0 + 32 + kr) * ATTN_DIM + c0 + nb, 0, 1);
#pragma unroll
      for (int j = 0; j < 4; ++j) {
        float4 f = src[j];
        Bs[nb + 4 * j    ][kr] = f2bf(f.x);
        Bs[nb + 4 * j + 1][kr] = f2bf(f.y);
        Bs[nb + 4 * j + 2][kr] = f2bf(f.z);
        Bs[nb + 4 * j + 3][kr] = f2bf(f.w);
      }
    }
    wait_async0();
    __syncthreads();

    Frag a;
#pragma unroll
    for (int s = 0; s < 8; ++s)
      a.u[s] = *(const unsigned int*)&As2[wave * 16 + ln][a_ksel(s) + hi * 8];
#pragma unroll
    for (int t = 0; t < 4; ++t) {
      Frag b;
#pragma unroll
      for (int s = 0; s < 8; ++s)
        b.u[s] = *(const unsigned int*)&Bs[t * 16 + ln][a_ksel(s) + hi * 8];
      acc[t] = wmma_bf16(a.v, b.v, acc[t]);
    }
    __syncthreads();
  }

#pragma unroll
  for (int t = 0; t < 4; ++t)
#pragma unroll
    for (int r = 0; r < 8; ++r) {
      int grow = row0 + wave * 16 + r + hi * 8;
      int gcol = c0 + t * 16 + ln;
      out[(size_t)grow * ATTN_DIM + gcol] = acc[t][r] + bo[gcol];
    }
}

// ---------------------------------------------------------------------------
extern "C" void kernel_launch(void* const* d_in, const int* in_sizes, int n_in,
                              void* d_out, int out_size, void* d_ws, size_t ws_size,
                              hipStream_t stream) {
  const float* x    = (const float*)d_in[0];
  const int*   mask = (const int*)  d_in[1];
  const float* Wq   = (const float*)d_in[2];
  const float* Wk   = (const float*)d_in[3];
  const float* Wv   = (const float*)d_in[4];
  const float* Wo   = (const float*)d_in[5];
  const float* bo   = (const float*)d_in[6];
  float* out = (float*)d_out;

  // workspace: 4 bf16 segments of B*H*N*D = 4,194,304 elements (32 MB total)
  const size_t SEG = (size_t)ATTN_B * ATTN_H * ATTN_N * ATTN_D;
  unsigned short* ws = (unsigned short*)d_ws;
  unsigned short* q  = ws;
  unsigned short* k  = ws + SEG;
  unsigned short* vt = ws + 2 * SEG;
  unsigned short* o  = ws + 3 * SEG;

  dim3 g1(48, 64, 1);                       // 3072/64 cols x 4096/64 rows
  qkv_proj_kernel<<<g1, 128, 0, stream>>>(x, Wq, Wk, Wv, q, k, vt);

  dim3 g2(ATTN_N / 64, ATTN_H, ATTN_B);     // 32 x 16 x 2
  attn_kernel<<<g2, 128, 0, stream>>>(q, k, vt, mask, o);

  dim3 g3(16, 64, 1);                       // 1024/64 cols x 4096/64 rows
  out_proj_kernel<<<g3, 128, 0, stream>>>(o, Wo, bo, out);
}
